// GraphSage_43654047596868
// MI455X (gfx1250) — matrix operand order repaired and checked by reading, not compiled, not run on hardware
//
#include <hip/hip_runtime.h>

// ---------------------------------------------------------------------------
// GraphSAGE on MI455X (gfx1250, wave32, WMMA).
// GEMMs run in f16 inputs / f32 accum via v_wmma_f32_16x16x32_f16.
// ---------------------------------------------------------------------------

typedef _Float16 v16h __attribute__((ext_vector_type(16)));
typedef _Float16 v8h  __attribute__((ext_vector_type(8)));
typedef _Float16 h4v  __attribute__((ext_vector_type(4)));
typedef _Float16 h2v  __attribute__((ext_vector_type(2)));
typedef float    v8f  __attribute__((ext_vector_type(8)));

#define NNODES   50000
#define DIM      512            // D_IN == D_OUT == 512
#define KDIM     1024           // 2 * DIM (self || agg)
#define FLATN    (NNODES * DIM) // 25,600,000
#define FC_BLOCKS 2048

// ---------------------------------------------------------------------------
// fp32 -> fp16 conversion (for W1 / W2), float4-granular.
// ---------------------------------------------------------------------------
__global__ void f32_to_f16_kernel(const float* __restrict__ src,
                                  _Float16* __restrict__ dst, int n4) {
  int i = blockIdx.x * blockDim.x + threadIdx.x;
  if (i < n4) {
    float4 v = reinterpret_cast<const float4*>(src)[i];
    h4v o = { (_Float16)v.x, (_Float16)v.y, (_Float16)v.z, (_Float16)v.w };
    reinterpret_cast<h4v*>(dst)[i] = o;
  }
}

// ---------------------------------------------------------------------------
// prep1: A1[n, 0:512] = f16(x[n]); A1[n, 512:1024] = f16(mean(x[nb0], x[nb1]))
// One block per node, 256 threads, 2 columns per thread.
// ---------------------------------------------------------------------------
__global__ __launch_bounds__(256)
void prep1_kernel(const float* __restrict__ x, const int* __restrict__ nb,
                  _Float16* __restrict__ A) {
  const int n = blockIdx.x;
  const int c = threadIdx.x * 2;
  const int i0 = nb[n * 2 + 0];
  const int i1 = nb[n * 2 + 1];
  float2 xs = *reinterpret_cast<const float2*>(x + (size_t)n  * DIM + c);
  float2 a0 = *reinterpret_cast<const float2*>(x + (size_t)i0 * DIM + c);
  float2 a1 = *reinterpret_cast<const float2*>(x + (size_t)i1 * DIM + c);
  _Float16* dst = A + (size_t)n * KDIM;
  h2v self = { (_Float16)xs.x, (_Float16)xs.y };
  h2v agg  = { (_Float16)(0.5f * (a0.x + a1.x)), (_Float16)(0.5f * (a0.y + a1.y)) };
  *reinterpret_cast<h2v*>(dst + c)       = self;
  *reinterpret_cast<h2v*>(dst + DIM + c) = agg;
}

// ---------------------------------------------------------------------------
// prep2: same, but source activations are already f16 [N, 512].
// ---------------------------------------------------------------------------
__global__ __launch_bounds__(256)
void prep2_kernel(const _Float16* __restrict__ h, const int* __restrict__ nb,
                  _Float16* __restrict__ A) {
  const int n = blockIdx.x;
  const int c = threadIdx.x * 2;
  const int i0 = nb[n * 2 + 0];
  const int i1 = nb[n * 2 + 1];
  h2v hs = *reinterpret_cast<const h2v*>(h + (size_t)n  * DIM + c);
  h2v n0 = *reinterpret_cast<const h2v*>(h + (size_t)i0 * DIM + c);
  h2v n1 = *reinterpret_cast<const h2v*>(h + (size_t)i1 * DIM + c);
  _Float16* dst = A + (size_t)n * KDIM;
  h2v agg = { (_Float16)(0.5f * ((float)n0[0] + (float)n1[0])),
              (_Float16)(0.5f * ((float)n0[1] + (float)n1[1])) };
  *reinterpret_cast<h2v*>(dst + c)       = hs;
  *reinterpret_cast<h2v*>(dst + DIM + c) = agg;
}

// ---------------------------------------------------------------------------
// WMMA GEMM + ReLU:  out[m,n] = relu( sum_k A[m,k] * W[n,k] )
//   A: [M, 1024] f16 row-major, W: [512, 1024] f16 row-major (acts as B = W^T)
// Block = 256 threads = 8 waves. Each block: 16 rows x 512 cols.
// A tile (16 x 1024 f16 = 32KB) staged once in LDS (padded stride to avoid
// 64-bank conflicts), each wave computes a 16x64 slice (4 accumulators,
// A-fragment reused across 4 WMMAs per k-step). 50000 = 3125 * 16 exactly.
//
// CDNA5 fragment layouts (ISA 7.12.2):
//   A (16-bit 16x32): lane l -> row l&15; halves 0..7 = K kb..kb+7,
//                     halves 8..15 = K kb+16..kb+23, kb = (l>>4)*8.
//   B (16-bit 32x16): lane l -> col l&15; halves 0..7 = K kb..kb+7, 8..15 = K kb+16..kb+23 (as A).
//   C/D f32:          lane l -> col l&15; vgpr v -> row v + 8*(l>>4).
// ---------------------------------------------------------------------------
template <int OUT_F16>
__global__ __launch_bounds__(256, 2)
void sage_gemm_relu_kernel(const _Float16* __restrict__ A,
                           const _Float16* __restrict__ W,
                           void* __restrict__ outp) {
  constexpr int LDSW = KDIM + 8;              // 1032 halves: +16B row pad
  __shared__ _Float16 ldsA[16 * LDSW];

  const int tid  = threadIdx.x;
  const int lane = tid & 31;
  const int wave = tid >> 5;                  // 0..7
  const int rowBlock = blockIdx.x * 16;

  // Stage the 16 x 1024 A tile into LDS. Tile is one contiguous block in
  // global (row stride == KDIM), copy in 16B chunks, coalesced.
  {
    const uint4* src = reinterpret_cast<const uint4*>(A + (size_t)rowBlock * KDIM);
#pragma unroll
    for (int it = 0; it < 8; ++it) {
      int c    = it * 256 + tid;              // 2048 chunks of 8 halves
      int r    = c >> 7;                      // 128 chunks per row
      int col8 = c & 127;
      uint4 v = src[r * 128 + col8];
      *reinterpret_cast<uint4*>(&ldsA[r * LDSW + col8 * 8]) = v;
    }
  }
  __syncthreads();

  const int lrow = lane & 15;
  const int hi   = lane >> 4;                 // 0 or 1
  const int kbA  = hi * 8;                    // A-fragment K chunk base
  const int kbB  = hi * 8;                    // B-fragment K chunk base
  const int colBase = wave * 64;

  v8f acc[4] = {};

  for (int kk = 0; kk < KDIM; kk += 32) {
    // A fragment from LDS: two 16B reads.
    const _Float16* ap = &ldsA[lrow * LDSW + kk + kbA];
    v8h alo = *reinterpret_cast<const v8h*>(ap);
    v8h ahi = *reinterpret_cast<const v8h*>(ap + 16);
    v16h afrag = __builtin_shufflevector(alo, ahi,
        0, 1, 2, 3, 4, 5, 6, 7, 8, 9, 10, 11, 12, 13, 14, 15);

#pragma unroll
    for (int c = 0; c < 4; ++c) {
      // B fragment: W row (colBase+c*16+lrow), halves K kb..kb+7 and kb+16..kb+23.
      const _Float16* bp =
          W + (size_t)(colBase + c * 16 + lrow) * KDIM + kk + kbB;
      v8h blo = *reinterpret_cast<const v8h*>(bp);
      v8h bhi = *reinterpret_cast<const v8h*>(bp + 16);
      v16h bfrag = __builtin_shufflevector(blo, bhi,
          0, 1, 2, 3, 4, 5, 6, 7, 8, 9, 10, 11, 12, 13, 14, 15);
#if defined(__gfx1250__)
      acc[c] = __builtin_amdgcn_wmma_f32_16x16x32_f16(
          /*neg_a=*/false, afrag, /*neg_b=*/false, bfrag,
          /*c_mod=*/(short)0, acc[c], /*reuse_a=*/false, /*reuse_b=*/false);
#endif
    }
  }

  // ReLU + store. lane l holds col colBase+c*16+(l&15), rows rowBlock+8*hi+v.
  const int outRow0 = rowBlock + hi * 8;
#pragma unroll
  for (int c = 0; c < 4; ++c) {
    const int n = colBase + c * 16 + lrow;
#pragma unroll
    for (int v = 0; v < 8; ++v) {
      float val = acc[c][v];
      val = val > 0.0f ? val : 0.0f;
      size_t idx = (size_t)(outRow0 + v) * DIM + n;
      if (OUT_F16) reinterpret_cast<_Float16*>(outp)[idx] = (_Float16)val;
      else         reinterpret_cast<float*>(outp)[idx]    = val;
    }
  }
}

// ---------------------------------------------------------------------------
// fc pass 1: per-block partial dot products of flat (25.6M f32) against the
// two fc_w rows. Deterministic tree reduction (no atomics -> replay-safe).
// ---------------------------------------------------------------------------
__global__ __launch_bounds__(256)
void fc_partial_kernel(const float* __restrict__ flat,
                       const float* __restrict__ fcw,
                       float* __restrict__ partial) {
  const int total4 = FLATN / 4;
  float s0 = 0.0f, s1 = 0.0f;
  const float4* f4  = reinterpret_cast<const float4*>(flat);
  const float4* w04 = reinterpret_cast<const float4*>(fcw);
  const float4* w14 = reinterpret_cast<const float4*>(fcw + FLATN);
  for (int i = blockIdx.x * blockDim.x + threadIdx.x; i < total4;
       i += gridDim.x * blockDim.x) {
    float4 h  = f4[i];
    float4 a = w04[i];
    float4 b = w14[i];
    s0 += h.x * a.x + h.y * a.y + h.z * a.z + h.w * a.w;
    s1 += h.x * b.x + h.y * b.y + h.z * b.z + h.w * b.w;
  }
  __shared__ float red[512];
  red[threadIdx.x]       = s0;
  red[256 + threadIdx.x] = s1;
  __syncthreads();
  for (int off = 128; off > 0; off >>= 1) {
    if (threadIdx.x < off) {
      red[threadIdx.x]       += red[threadIdx.x + off];
      red[256 + threadIdx.x] += red[256 + threadIdx.x + off];
    }
    __syncthreads();
  }
  if (threadIdx.x == 0) {
    partial[blockIdx.x]             = red[0];
    partial[FC_BLOCKS + blockIdx.x] = red[256];
  }
}

// ---------------------------------------------------------------------------
// fc pass 2: final reduce + bias + log_softmax -> d_out[FLATN .. FLATN+1].
// ---------------------------------------------------------------------------
__global__ __launch_bounds__(256)
void fc_final_kernel(const float* __restrict__ partial,
                     const float* __restrict__ fcb,
                     float* __restrict__ out2) {
  float s0 = 0.0f, s1 = 0.0f;
  for (int i = threadIdx.x; i < FC_BLOCKS; i += 256) {
    s0 += partial[i];
    s1 += partial[FC_BLOCKS + i];
  }
  __shared__ float red[512];
  red[threadIdx.x]       = s0;
  red[256 + threadIdx.x] = s1;
  __syncthreads();
  for (int off = 128; off > 0; off >>= 1) {
    if (threadIdx.x < off) {
      red[threadIdx.x]       += red[threadIdx.x + off];
      red[256 + threadIdx.x] += red[256 + threadIdx.x + off];
    }
    __syncthreads();
  }
  if (threadIdx.x == 0) {
    float l0 = red[0]   + fcb[0];
    float l1 = red[256] + fcb[1];
    float m  = fmaxf(l0, l1);
    float lse = m + logf(expf(l0 - m) + expf(l1 - m));
    out2[0] = l0 - lse;
    out2[1] = l1 - lse;
  }
}

// ---------------------------------------------------------------------------
// Launch. Workspace layout (bytes, all 256-aligned):
//   [0,           102,400,000)  A-buffer f16 [N,1024]  (reused: A1 then A2)
//   [102,400,000, 153,600,000)  h1 f16 [N,512]
//   [153,600,000, 154,648,576)  W1 f16
//   [154,648,576, 155,697,152)  W2 f16
//   [155,697,152, +16,384)      fc partials
// ---------------------------------------------------------------------------
extern "C" void kernel_launch(void* const* d_in, const int* in_sizes, int n_in,
                              void* d_out, int out_size, void* d_ws,
                              size_t ws_size, hipStream_t stream) {
  const float* x   = (const float*)d_in[0];
  const int*   nb1 = (const int*)d_in[1];
  const int*   nb2 = (const int*)d_in[2];
  const float* W1  = (const float*)d_in[3];
  const float* W2  = (const float*)d_in[4];
  const float* fcw = (const float*)d_in[5];
  const float* fcb = (const float*)d_in[6];
  float* out = (float*)d_out;

  char* ws = (char*)d_ws;
  _Float16* Abuf = (_Float16*)(ws);
  _Float16* h1f  = (_Float16*)(ws + 102400000);
  _Float16* W1f  = (_Float16*)(ws + 153600000);
  _Float16* W2f  = (_Float16*)(ws + 154648576);
  float* partial = (float*)(ws + 155697152);

  // Weight conversions: 512*1024 = 524,288 floats = 131,072 float4s each.
  f32_to_f16_kernel<<<512, 256, 0, stream>>>(W1, W1f, 131072);
  f32_to_f16_kernel<<<512, 256, 0, stream>>>(W2, W2f, 131072);

  // Layer 1
  prep1_kernel<<<NNODES, 256, 0, stream>>>(x, nb1, Abuf);
  sage_gemm_relu_kernel<1><<<NNODES / 16, 256, 0, stream>>>(Abuf, W1f, h1f);

  // Layer 2 (writes fp32 activations straight into d_out[0 .. FLATN))
  prep2_kernel<<<NNODES, 256, 0, stream>>>(h1f, nb2, Abuf);
  sage_gemm_relu_kernel<0><<<NNODES / 16, 256, 0, stream>>>(Abuf, W2f, out);

  // Classifier + log_softmax -> d_out[FLATN], d_out[FLATN+1]
  fc_partial_kernel<<<FC_BLOCKS, 256, 0, stream>>>(out, fcw, partial);
  fc_final_kernel<<<1, 256, 0, stream>>>(partial, fcb, out + FLATN);
}
